// FlashSigmoidAttention_5085241278761
// MI455X (gfx1250) — compile-verified, matrix operand order checked by reading, not run on hardware
//
#include <hip/hip_runtime.h>
#include <hip/hip_bf16.h>
#include <stdint.h>

// ---------------------------------------------------------------------------
// MI455X / gfx1250 flash sigmoid attention, bf16 WMMA (f32 accum) everywhere.
//   B=2, N=2048, C=2048, H=16, D=128.  All GEMMs are C = A * B^T, row-major.
// Round 3: widened operand working sets so the scheduler can clause loads and
// stagger s_wait_loadcnt instead of load->wait(0)->wmma serialization:
//   GEMM  : 32x64/wave; 6 operands loaded up front per k-step, then 8 WMMAs
//   Attn  : 32 q-rows/wave; P*V consumes V in groups of 4 operands
// ---------------------------------------------------------------------------

typedef __bf16    v16bf __attribute__((ext_vector_type(16)));
typedef float     v8f   __attribute__((ext_vector_type(8)));
typedef uint32_t  u32x4 __attribute__((ext_vector_type(4)));

#define SIG_BIAS (-8.0f)

// round-to-nearest-even f32 -> bf16 bits
static __device__ __forceinline__ uint16_t f2bf(float f) {
  uint32_t u = __builtin_bit_cast(uint32_t, f);
  uint32_t r = u + 0x7FFFu + ((u >> 16) & 1u);
  return (uint16_t)(r >> 16);
}

static __device__ __forceinline__ v8f v8f_zero() {
  v8f z;
#pragma unroll
  for (int i = 0; i < 8; ++i) z[i] = 0.0f;
  return z;
}

// Load one 16x16x32 bf16 WMMA A/B operand slice for this lane:
// two contiguous 8-element (16-byte) chunks. Callers pass
//   p0 = row_base + k0 + 8*(lane>=16),  p1 = p0 + 16.
static __device__ __forceinline__ v16bf load_ab(const uint16_t* p0, const uint16_t* p1) {
  union { v16bf v; u32x4 q[2]; } u;
  u.q[0] = *reinterpret_cast<const u32x4*>(p0);
  u.q[1] = *reinterpret_cast<const u32x4*>(p1);
  return u.v;
}

static __device__ __forceinline__ v8f wmma_bf16(v16bf a, v16bf b, v8f c) {
  return __builtin_amdgcn_wmma_f32_16x16x32_bf16(false, a, false, b, (short)0, c,
                                                 false, false);
}

// ---------------------------------------------------------------------------
// fp32 -> bf16 bulk convert
// ---------------------------------------------------------------------------
__global__ void cvt_f32_bf16(const float* __restrict__ src,
                             uint16_t* __restrict__ dst, int n) {
  int i = blockIdx.x * blockDim.x + threadIdx.x;
  int stride = gridDim.x * blockDim.x;
  for (; i < n; i += stride) dst[i] = f2bf(src[i]);
}

// ---------------------------------------------------------------------------
// GEMM: C[M x Nout] = A[M x K] * Bm[Nout x K]^T, bf16 in, f32 accum.
// One wave owns a 32 x 64 output tile (2 row-subtiles x 4 col-subtiles).
// MODE 0: QKV epilogue -> scatter into qb/kb (scaled) and vtb (transposed).
// MODE 1: proj epilogue -> f32 store to outf.
// ---------------------------------------------------------------------------
template <int MODE>
__global__ __launch_bounds__(256) void gemm_bt(
    const uint16_t* __restrict__ A, const uint16_t* __restrict__ Bm,
    int M, int Nout, int K, float* __restrict__ outf,
    uint16_t* __restrict__ qb, uint16_t* __restrict__ kb,
    uint16_t* __restrict__ vtb, float qk_scale) {
  const int lane = threadIdx.x & 31;
  const int wv   = threadIdx.x >> 5;
  const int wid  = blockIdx.x * 8 + wv;
  const int nt4n = Nout >> 6;
  const int mt   = wid / nt4n;
  const int nt4  = wid % nt4n;
  if (mt * 32 >= M) return;  // wave-uniform

  const int lm   = lane & 15;
  const int half = lane >> 4;

  const uint16_t* arow0 = A  + (size_t)(mt * 32 + lm) * K + half * 8;
  const uint16_t* arow1 = arow0 + (size_t)16 * K;
  const uint16_t* brow0 = Bm + (size_t)(nt4 * 64 + lm) * K + half * 8;

  v8f acc[2][4];
#pragma unroll
  for (int i = 0; i < 2; ++i)
#pragma unroll
    for (int j = 0; j < 4; ++j) acc[i][j] = v8f_zero();

  for (int k0 = 0; k0 < K; k0 += 32) {
    // bind all 6 operands first -> one long load clause, staggered waits
    v16bf a0 = load_ab(arow0 + k0, arow0 + k0 + 16);
    v16bf a1 = load_ab(arow1 + k0, arow1 + k0 + 16);
    v16bf b[4];
#pragma unroll
    for (int j = 0; j < 4; ++j) {
      const uint16_t* br = brow0 + (size_t)(j * 16) * K;
      b[j] = load_ab(br + k0, br + k0 + 16);
    }
#pragma unroll
    for (int j = 0; j < 4; ++j) {
      acc[0][j] = wmma_bf16(a0, b[j], acc[0][j]);
      acc[1][j] = wmma_bf16(a1, b[j], acc[1][j]);
    }
  }

  // D layout: lane -> col = 64*nt4 + 16*j + lm ;
  //           VGPR r -> row = 32*mt + 16*i + r + 8*half
  if (MODE == 0) {
#pragma unroll
    for (int j = 0; j < 4; ++j) {
      const int e = nt4 * 64 + j * 16 + lm;     // output channel in [0, 3C)
      const int s = e >> 11;                    // 0=q, 1=k, 2=v
      const int rem = e & 2047;
      const int h = rem >> 7;
      const int d = rem & 127;
#pragma unroll
      for (int i = 0; i < 2; ++i) {
#pragma unroll
        for (int r = 0; r < 8; ++r) {
          const int m = mt * 32 + i * 16 + r + half * 8;  // b*N + n
          const int bb = m >> 11;
          const int n = m & 2047;
          const size_t bh = (size_t)(bb * 16 + h);
          const float val = acc[i][j][r];
          if (s == 0) {
            qb[bh * 262144 + (size_t)n * 128 + d] = f2bf(val * qk_scale);
          } else if (s == 1) {
            kb[bh * 262144 + (size_t)n * 128 + d] = f2bf(val * qk_scale);
          } else {
            vtb[bh * 262144 + (size_t)d * 2048 + n] = f2bf(val);  // V transposed
          }
        }
      }
    }
  } else {
#pragma unroll
    for (int j = 0; j < 4; ++j) {
      const int col = nt4 * 64 + j * 16 + lm;
#pragma unroll
      for (int i = 0; i < 2; ++i) {
#pragma unroll
        for (int r = 0; r < 8; ++r) {
          const int m = mt * 32 + i * 16 + r + half * 8;
          outf[(size_t)m * Nout + col] = acc[i][j][r];
        }
      }
    }
  }
}

// ---------------------------------------------------------------------------
// Flash sigmoid attention. One wave owns a 32-row q-tile of one (b,h),
// processed as two 16-row subtiles t=0,1 that share all K/V operand loads.
// S^T = K * Q^T is computed so the sigmoid'd probabilities land directly in
// A-operand layout for the P*V WMMA (no cross-lane transpose).
// ---------------------------------------------------------------------------
__global__ __launch_bounds__(256) void attn_kernel(
    const uint16_t* __restrict__ qb, const uint16_t* __restrict__ kb,
    const uint16_t* __restrict__ vtb, float* __restrict__ out_av,
    uint16_t* __restrict__ avb) {
  const int lane = threadIdx.x & 31;
  const int wv   = threadIdx.x >> 5;
  const int qt   = blockIdx.x * 8 + wv;   // 0..63
  const int bh   = blockIdx.y;            // 0..31
  const int lm   = lane & 15;
  const int half = lane >> 4;
  const int q0   = qt * 32;
  const int b    = bh >> 4;
  const int h    = bh & 15;

  const uint16_t* qbase = qb  + (size_t)bh * 262144;
  const uint16_t* kbase = kb  + (size_t)bh * 262144;
  const uint16_t* vbase = vtb + (size_t)bh * 262144;

  // Q tiles, resident: double as B-operands (Q^T) for S^T = K * Q^T.
  v16bf Qa[2][4];
#pragma unroll
  for (int t = 0; t < 2; ++t) {
#pragma unroll
    for (int dc = 0; dc < 4; ++dc) {
      const uint16_t* p =
          qbase + (size_t)(q0 + t * 16 + lm) * 128 + dc * 32 + half * 8;
      Qa[t][dc] = load_ab(p, p + 16);
    }
  }

  v8f o[2][8];
#pragma unroll
  for (int t = 0; t < 2; ++t)
#pragma unroll
    for (int c = 0; c < 8; ++c) o[t][c] = v8f_zero();

  const int qr0 = q0 + lm;        // q index of this lane in subtile 0
  const int qr1 = q0 + 16 + lm;   // q index of this lane in subtile 1

  for (int j0 = 0; j0 <= q0; j0 += 32) {   // causal: kv tiles of 32
    // st[t][k2]: S^T for q-subtile t, k rows j0+16*k2 .. +15
    v8f st[2][2];
#pragma unroll
    for (int t = 0; t < 2; ++t)
#pragma unroll
      for (int k2 = 0; k2 < 2; ++k2) st[t][k2] = v8f_zero();

#pragma unroll
    for (int k2 = 0; k2 < 2; ++k2) {
      // bind all 4 K operands for this 16-row block, then 8 WMMAs
      v16bf ka[4];
#pragma unroll
      for (int dc = 0; dc < 4; ++dc) {
        const uint16_t* kp =
            kbase + (size_t)(j0 + k2 * 16 + lm) * 128 + dc * 32 + half * 8;
        ka[dc] = load_ab(kp, kp + 16);
      }
#pragma unroll
      for (int dc = 0; dc < 4; ++dc) {
        st[0][k2] = wmma_bf16(ka[dc], Qa[0][dc], st[0][k2]);
        st[1][k2] = wmma_bf16(ka[dc], Qa[1][dc], st[1][k2]);
      }
    }

    // bias + sigmoid + causal mask, pack straight into A-operand layout
    union { v16bf v; uint16_t u[16]; } pa[2];
#pragma unroll
    for (int r = 0; r < 8; ++r) {
      const int k1 = j0 + r + half * 8;
      const int k2i = k1 + 16;
#pragma unroll
      for (int t = 0; t < 2; ++t) {
        const int qrow = (t == 0) ? qr0 : qr1;
        const float t1 = st[t][0][r] + SIG_BIAS;
        const float p1 = (k1 <= qrow) ? 1.0f / (1.0f + __expf(-t1)) : 0.0f;
        pa[t].u[r] = f2bf(p1);
        const float t2 = st[t][1][r] + SIG_BIAS;
        const float p2 = (k2i <= qrow) ? 1.0f / (1.0f + __expf(-t2)) : 0.0f;
        pa[t].u[8 + r] = f2bf(p2);
      }
    }

    // O += P * V : V operands (transposed V, row-major in d) consumed in two
    // groups of 4 so loads clause up and waits stagger; shared by both
    // q-subtiles.
#pragma unroll
    for (int cg = 0; cg < 2; ++cg) {
      v16bf vb[4];
#pragma unroll
      for (int c4 = 0; c4 < 4; ++c4) {
        const int c = cg * 4 + c4;
        const uint16_t* vp =
            vbase + (size_t)(c * 16 + lm) * 2048 + j0 + half * 8;
        vb[c4] = load_ab(vp, vp + 16);
      }
#pragma unroll
      for (int c4 = 0; c4 < 4; ++c4) {
        const int c = cg * 4 + c4;
        o[0][c] = wmma_bf16(pa[0].v, vb[c4], o[0][c]);
        o[1][c] = wmma_bf16(pa[1].v, vb[c4], o[1][c]);
      }
    }
  }

  // write attn_times_v (fp32 to d_out, bf16 copy for the proj GEMM)
#pragma unroll
  for (int t = 0; t < 2; ++t) {
#pragma unroll
    for (int c = 0; c < 8; ++c) {
      const int d = c * 16 + lm;
#pragma unroll
      for (int r = 0; r < 8; ++r) {
        const int q = q0 + t * 16 + r + half * 8;
        const size_t idx = ((size_t)b * 2048 + q) * 2048 + h * 128 + d;
        const float val = o[t][c][r];
        out_av[idx] = val;
        avb[idx] = f2bf(val);
      }
    }
  }
}

// ---------------------------------------------------------------------------
// launch
// ---------------------------------------------------------------------------
extern "C" void kernel_launch(void* const* d_in, const int* in_sizes, int n_in,
                              void* d_out, int out_size, void* d_ws, size_t ws_size,
                              hipStream_t stream) {
  (void)in_sizes; (void)n_in; (void)out_size; (void)ws_size;
  const float* x     = (const float*)d_in[0];   // [2,2048,2048]
  const float* wqkv  = (const float*)d_in[1];   // [6144,2048]
  const float* wproj = (const float*)d_in[2];   // [2048,2048]

  float* out_av   = (float*)d_out;                          // [2,2048,2048]
  float* out_proj = out_av + (size_t)2 * 2048 * 2048;       // [2,2048,2048]

  uint8_t* ws = (uint8_t*)d_ws;
  uint16_t* xb     = (uint16_t*)(ws);                       // 16 MiB
  uint16_t* wqkvb  = (uint16_t*)(ws + 16777216ull);         // 24 MiB
  uint16_t* wprojb = (uint16_t*)(ws + 41943040ull);         //  8 MiB
  uint16_t* qb     = (uint16_t*)(ws + 50331648ull);         // 16 MiB  [bh][n][d]
  uint16_t* kb     = (uint16_t*)(ws + 67108864ull);         // 16 MiB  [bh][n][d]
  uint16_t* vtb    = (uint16_t*)(ws + 83886080ull);         // 16 MiB  [bh][d][n]
  uint16_t* avb    = (uint16_t*)(ws + 100663296ull);        // 16 MiB

  const float qk_scale = 0.29730177875068026f;  // 128^(-0.25)

  cvt_f32_bf16<<<2048, 256, 0, stream>>>(x, xb, 8388608);
  cvt_f32_bf16<<<2048, 256, 0, stream>>>(wqkv, wqkvb, 12582912);
  cvt_f32_bf16<<<1024, 256, 0, stream>>>(wproj, wprojb, 4194304);

  // QKV: M=4096, Nout=6144, K=2048 -> (4096/32)*(6144/64) = 12288 waves
  gemm_bt<0><<<1536, 256, 0, stream>>>(xb, wqkvb, 4096, 6144, 2048,
                                       nullptr, qb, kb, vtb, qk_scale);

  // attention: 64 q-tiles (32 rows each) x 32 (b,h)
  attn_kernel<<<dim3(8, 32), 256, 0, stream>>>(qb, kb, vtb, out_av, avb);

  // proj: M=4096, Nout=2048, K=2048 -> (4096/32)*(2048/64) = 4096 waves
  gemm_bt<1><<<512, 256, 0, stream>>>(avb, wprojb, 4096, 2048, 2048,
                                      out_proj, nullptr, nullptr, nullptr, 0.0f);
}
